// ProtGramDirectGCN_19018115186741
// MI455X (gfx1250) — matrix operand order repaired and checked by reading, not compiled, not run on hardware
//
#include <hip/hip_runtime.h>
#include <hip/hip_bf16.h>
#include <cstddef>
#include <cstdint>

// ---------------------------------------------------------------------------
// Types for CDNA5 WMMA (wave32, 16x16x32 bf16 -> f32 accum)
// ---------------------------------------------------------------------------
typedef __bf16 bf16_t;
typedef __attribute__((ext_vector_type(16))) __bf16 v16bf;
typedef __attribute__((ext_vector_type(8)))  __bf16 v8bf;
typedef __attribute__((ext_vector_type(8)))  float  v8f;

// ---------------------------------------------------------------------------
// Input index map (recursive insertion-order flatten of setup_inputs dict;
// None entries skipped).
// ---------------------------------------------------------------------------
#define IN_X        0
#define IN_EI_IN    1
#define IN_EW_IN    2
#define IN_EI_OUT   3
#define IN_EW_OUT   4
#define IN_EI_UN    5
#define IN_EW_UN    6
#define IN_ORIG     7
#define IN_L0       8   // Wmi,Wmo,Wun,Wsh,b_mi,b_mo,b_un,b_sh_in,b_sh_out,b_un_sh,C_in,C_out,C_dir,C_und,C_all,const
#define IN_L1       24
#define IN_RES_W    40
#define IN_RES_B    41
#define IN_DEC_W1   42
#define IN_DEC_B1   43
#define IN_DEC_W2   44
#define IN_DEC_B2   45
#define IN_PE       46
// offsets inside a layer block
#define L_WMI 0
#define L_WMO 1
#define L_WUN 2
#define L_WSH 3
#define L_BMI 4
#define L_BMO 5
#define L_BUN 6
#define L_BSHI 7
#define L_BSHO 8
#define L_BUNSH 9
#define L_CIN 10
#define L_COUT 11
#define L_CDIR 12
#define L_CUND 13
#define L_CALL 14
#define L_CONST 15

// ---------------------------------------------------------------------------
// Utility kernels
// ---------------------------------------------------------------------------
__global__ void fill_f32(float* __restrict__ p, long long n, float v) {
  long long i = (long long)blockIdx.x * blockDim.x + threadIdx.x;
  long long stride = (long long)gridDim.x * blockDim.x;
  for (; i < n; i += stride) p[i] = v;
}

// dst = bf16(a + b) (b optional)
__global__ void addcvt_bf16(bf16_t* __restrict__ dst, const float* __restrict__ a,
                            const float* __restrict__ b, int n) {
  int i = blockIdx.x * blockDim.x + threadIdx.x;
  if (i < n) {
    float v = a[i] + (b ? b[i] : 0.0f);
    dst[i] = (bf16_t)v;
  }
}

// h = x + pe (NGRAM=2, ONEG=64: pe flat index == channel index), also bf16 copy
__global__ void pe_add(const float* __restrict__ x, const float* __restrict__ pe,
                       float* __restrict__ h, bf16_t* __restrict__ hb) {
  int n = blockIdx.x; int c = threadIdx.x;        // blockDim.x == 128
  size_t i = (size_t)n * 128 + c;
  float v = x[i] + pe[c];
  h[i] = v;
  hb[i] = (bf16_t)v;
}

// ---------------------------------------------------------------------------
// bf16 WMMA GEMM:  out[N,Cout] = A[N,K](bf16) @ W[Cout,K]^T(bf16) (+bias,relu)
// One wave32 per 16x16 output tile, K stepped by 32 through
// v_wmma_f32_16x16x32_bf16. A/B use contiguous k order (k = 16h + s); C/D layout per CDNA5 ISA 7.12.2.
// ---------------------------------------------------------------------------
__global__ __launch_bounds__(256)
void gemm_bf16_wmma(const bf16_t* __restrict__ A, const bf16_t* __restrict__ W,
                    const float* __restrict__ bias, float* __restrict__ outF,
                    bf16_t* __restrict__ outB, int Nrows, int K, int Cout, int relu)
{
  const int wavesPerBlock = blockDim.x >> 5;
  const int wid  = blockIdx.x * wavesPerBlock + ((int)threadIdx.x >> 5);
  const int lane = (int)threadIdx.x & 31;
  const int tilesN = Cout >> 4;
  const int totalTiles = (Nrows >> 4) * tilesN;
  if (wid >= totalTiles) return;
  const int tm = (wid / tilesN) << 4;
  const int tn = (wid % tilesN) << 4;

  const int halfSel = lane >> 4;      // lane group 0..15 vs 16..31
  const int l15     = lane & 15;
  const int rowA = tm + l15;          // A: one matrix row per lane (dup over halves)
  const int colB = tn + l15;          // B = W^T: one output column per lane

  v8f acc = {};
  for (int kk = 0; kk < K; kk += 32) {
    // A frag (16x32 bf16): lanes 0-15 carry K= kk+[0..15]
    //                      lanes 16-31 carry K= kk+[16..31]
    const bf16_t* ap = A + (size_t)rowA * K + kk + (halfSel << 4);
    v8bf alo = *(const v8bf*)(ap);
    v8bf ahi = *(const v8bf*)(ap + 8);
    v16bf af = __builtin_shufflevector(alo, ahi,
        0,1,2,3,4,5,6,7, 8,9,10,11,12,13,14,15);
    // B frag (32x16 bf16): lanes 0-15 carry K=kk+[0..15], lanes 16-31 K=kk+[16..31]
    const bf16_t* bp = W + (size_t)colB * K + kk + (halfSel << 4);
    v16bf bfr = *(const v16bf*)(bp);
    acc = __builtin_amdgcn_wmma_f32_16x16x32_bf16(
        /*neg_a=*/false, af, /*neg_b=*/false, bfr,
        /*c_mod=*/(short)0, acc, /*reuse_a=*/false, /*reuse_b=*/false);
  }

  // C/D layout: VGPR r holds M=r (lanes 0-15) / M=8+r (lanes 16-31), N=lane&15
  const int col = tn + l15;
  const float bval = bias ? bias[col] : 0.0f;
#pragma unroll
  for (int r = 0; r < 8; ++r) {
    int row = tm + r + (halfSel << 3);
    float v = acc[r] + bval;
    if (relu) v = fmaxf(v, 0.0f);
    size_t oi = (size_t)row * Cout + col;
    if (outF) outF[oi] = v;
    if (outB) outB[oi] = (bf16_t)v;
  }
}

// ---------------------------------------------------------------------------
// Edge scatter: agg[dst] += ew * t[src].  blockDim = (C/4, 256/(C/4)).
// Lanes of one "row" cover a full feature row (float4/lane) -> coalesced
// gathers; scatter via GLOBAL_ATOMIC_ADD_F32 (working set lives in 192MB L2).
// ---------------------------------------------------------------------------
__global__ void edge_prop(const float* __restrict__ t, const int* __restrict__ ei,
                          const float* __restrict__ ew, float* __restrict__ agg,
                          int E, int C) {
  int e = blockIdx.x * blockDim.y + threadIdx.y;
  if (e >= E) return;
  int q = threadIdx.x;                 // float4 group within the row
  int src = ei[e];
  int dst = ei[E + e];
  float w = ew[e];
  const float4 v = ((const float4*)(t + (size_t)src * C))[q];
  float* o = agg + (size_t)dst * C + (q << 2);
  atomicAdd(o + 0, w * v.x);
  atomicAdd(o + 1, w * v.y);
  atomicAdd(o + 2, w * v.z);
  atomicAdd(o + 3, w * v.w);
}

// ---------------------------------------------------------------------------
// Fused combine: biases + per-node coefficients + const + residual + leakyReLU
// writes next h (f32) and its bf16 mirror for the next WMMA stage.
// ---------------------------------------------------------------------------
__global__ void combine_kernel(
    const float* __restrict__ aggIn, const float* __restrict__ aggOut, const float* __restrict__ aggUn,
    const float* __restrict__ b_mi, const float* __restrict__ b_sh_in,
    const float* __restrict__ b_mo, const float* __restrict__ b_sh_out,
    const float* __restrict__ b_un, const float* __restrict__ b_un_sh,
    const float* __restrict__ Cin, const float* __restrict__ CoutA,
    const float* __restrict__ Cdir, const float* __restrict__ Cund,
    const float* __restrict__ Call, const float* __restrict__ cons,
    const int* __restrict__ orig_idx, const float* __restrict__ res,
    float* __restrict__ hOut, bf16_t* __restrict__ hbOut, int C)
{
  int n = blockIdx.x; int c = threadIdx.x;        // blockDim.x == C
  int o = orig_idx[n];
  size_t i = (size_t)n * C + c;
  float ic = aggIn[i]  + b_mi[c] + b_sh_in[c];
  float oc = aggOut[i] + b_mo[c] + b_sh_out[c];
  float uc = aggUn[i]  + b_un[c] + b_un_sh[c];
  float gcn = Call[o] * (Cund[o] * uc + Cdir[o] * (Cin[o] * ic + CoutA[o] * oc))
            + cons[(size_t)o * C + c];
  float v = gcn + res[i];
  v = (v >= 0.0f) ? v : 0.01f * v;                // leaky_relu(0.01)
  hOut[i] = v;
  hbOut[i] = (bf16_t)v;
}

// ---------------------------------------------------------------------------
// log_softmax over exactly 32 classes: one wave32 per row, lane == class
// ---------------------------------------------------------------------------
__global__ void logsoftmax32(const float* __restrict__ logits,
                             float* __restrict__ out, int Nrows) {
  int row  = blockIdx.x * (blockDim.x >> 5) + ((int)threadIdx.x >> 5);
  int lane = (int)threadIdx.x & 31;
  if (row >= Nrows) return;
  float v = logits[(size_t)row * 32 + lane];
  float m = v;
#pragma unroll
  for (int off = 16; off; off >>= 1) m = fmaxf(m, __shfl_xor(m, off, 32));
  float s = expf(v - m);
#pragma unroll
  for (int off = 16; off; off >>= 1) s += __shfl_xor(s, off, 32);
  out[(size_t)row * 32 + lane] = v - m - logf(s);
}

// L2 normalize rows of 64: one wave32 per row, 2 elements per lane
__global__ void l2norm64(const float* __restrict__ h, float* __restrict__ out, int Nrows) {
  int row  = blockIdx.x * (blockDim.x >> 5) + ((int)threadIdx.x >> 5);
  int lane = (int)threadIdx.x & 31;
  if (row >= Nrows) return;
  float v0 = h[(size_t)row * 64 + lane];
  float v1 = h[(size_t)row * 64 + 32 + lane];
  float ss = v0 * v0 + v1 * v1;
#pragma unroll
  for (int off = 16; off; off >>= 1) ss += __shfl_xor(ss, off, 32);
  float nrm = fmaxf(sqrtf(ss), 1e-12f);
  out[(size_t)row * 64 + lane]      = v0 / nrm;
  out[(size_t)row * 64 + 32 + lane] = v1 / nrm;
}

// ---------------------------------------------------------------------------
// Host-side orchestration
// ---------------------------------------------------------------------------
static inline void launch_gemm(const bf16_t* A, const bf16_t* W, const float* bias,
                               float* outF, bf16_t* outB, int N, int K, int Cout,
                               int relu, hipStream_t s) {
  int tiles = (N >> 4) * (Cout >> 4);
  int wpb = 8;                                     // 8 waves / 256 threads
  int blocks = (tiles + wpb - 1) / wpb;
  gemm_bf16_wmma<<<blocks, wpb * 32, 0, s>>>(A, W, bias, outF, outB, N, K, Cout, relu);
}

static inline void launch_prop(const float* t, const int* ei, const float* ew,
                               float* agg, int E, int C, hipStream_t s) {
  int cg = C >> 2;
  dim3 block(cg, 256 / cg);
  int blocks = (E + (int)block.y - 1) / (int)block.y;
  edge_prop<<<blocks, block, 0, s>>>(t, ei, ew, agg, E, C);
}

extern "C" void kernel_launch(void* const* d_in, const int* in_sizes, int n_in,
                              void* d_out, int out_size, void* d_ws, size_t ws_size,
                              hipStream_t stream) {
  const int N = in_sizes[IN_X] / 128;
  const int E = in_sizes[IN_EW_IN];

  const float* x    = (const float*)d_in[IN_X];
  const int*   orig = (const int*)d_in[IN_ORIG];
  const float* pe   = (const float*)d_in[IN_PE];
  const int*   ei[3]  = { (const int*)d_in[IN_EI_IN], (const int*)d_in[IN_EI_OUT], (const int*)d_in[IN_EI_UN] };
  const float* ewp[3] = { (const float*)d_in[IN_EW_IN], (const float*)d_in[IN_EW_OUT], (const float*)d_in[IN_EW_UN] };

  // ---- carve workspace ----
  size_t off = 0;
  auto carve = [&](size_t bytes) -> void* {
    off = (off + 255) & ~(size_t)255;
    void* p = (char*)d_ws + off;
    off += bytes;
    return p;
  };
  float*  h    = (float*)carve((size_t)N * 128 * 4);
  bf16_t* hb   = (bf16_t*)carve((size_t)N * 128 * 2);
  float*  t    = (float*)carve((size_t)N * 128 * 4);
  float*  agg[3];
  for (int s = 0; s < 3; ++s) agg[s] = (float*)carve((size_t)N * 128 * 4);
  bf16_t* z1b  = (bf16_t*)carve((size_t)N * 32 * 2);
  bf16_t* W0c[3], *W1c[3];
  for (int s = 0; s < 3; ++s) W0c[s] = (bf16_t*)carve((size_t)128 * 128 * 2);
  for (int s = 0; s < 3; ++s) W1c[s] = (bf16_t*)carve((size_t)64 * 128 * 2);
  bf16_t* WresB = (bf16_t*)carve((size_t)64 * 128 * 2);
  bf16_t* dW1   = (bf16_t*)carve((size_t)32 * 64 * 2);
  bf16_t* dW2   = (bf16_t*)carve((size_t)32 * 32 * 2);
  (void)ws_size; (void)n_in;

  // ---- weight prep: W_combined = W_kind + W_sh (prop is linear) -> bf16 ----
  const int kindOf[3] = { L_WMI, L_WMO, L_WUN };
  for (int s = 0; s < 3; ++s) {
    addcvt_bf16<<<(128 * 128 + 255) / 256, 256, 0, stream>>>(
        W0c[s], (const float*)d_in[IN_L0 + kindOf[s]], (const float*)d_in[IN_L0 + L_WSH], 128 * 128);
    addcvt_bf16<<<(64 * 128 + 255) / 256, 256, 0, stream>>>(
        W1c[s], (const float*)d_in[IN_L1 + kindOf[s]], (const float*)d_in[IN_L1 + L_WSH], 64 * 128);
  }
  addcvt_bf16<<<(64 * 128 + 255) / 256, 256, 0, stream>>>(WresB, (const float*)d_in[IN_RES_W], nullptr, 64 * 128);
  addcvt_bf16<<<(32 * 64 + 255) / 256, 256, 0, stream>>>(dW1, (const float*)d_in[IN_DEC_W1], nullptr, 32 * 64);
  addcvt_bf16<<<(32 * 32 + 255) / 256, 256, 0, stream>>>(dW2, (const float*)d_in[IN_DEC_W2], nullptr, 32 * 32);

  // ---- positional encoding ----
  pe_add<<<N, 128, 0, stream>>>(x, pe, h, hb);

  // ---- two GCN layers ----
  for (int layer = 0; layer < 2; ++layer) {
    const int Cin  = 128;
    const int Cout = (layer == 0) ? 128 : 64;
    const int LB   = (layer == 0) ? IN_L0 : IN_L1;
    bf16_t** Wc    = (layer == 0) ? W0c : W1c;

    fill_f32<<<4096, 256, 0, stream>>>(agg[0], (long long)N * Cout, 0.0f);
    fill_f32<<<4096, 256, 0, stream>>>(agg[1], (long long)N * Cout, 0.0f);
    fill_f32<<<4096, 256, 0, stream>>>(agg[2], (long long)N * Cout, 0.0f);

    for (int s = 0; s < 3; ++s) {
      launch_gemm(hb, Wc[s], nullptr, t, nullptr, N, Cin, Cout, 0, stream);
      launch_prop(t, ei[s], ewp[s], agg[s], E, Cout, stream);
    }

    const float* res;
    if (layer == 0) {
      res = h;                                   // identity residual
    } else {
      launch_gemm(hb, WresB, (const float*)d_in[IN_RES_B], t, nullptr, N, 128, 64, 0, stream);
      res = t;                                   // projected residual
    }

    combine_kernel<<<N, Cout, 0, stream>>>(
        agg[0], agg[1], agg[2],
        (const float*)d_in[LB + L_BMI],  (const float*)d_in[LB + L_BSHI],
        (const float*)d_in[LB + L_BMO],  (const float*)d_in[LB + L_BSHO],
        (const float*)d_in[LB + L_BUN],  (const float*)d_in[LB + L_BUNSH],
        (const float*)d_in[LB + L_CIN],  (const float*)d_in[LB + L_COUT],
        (const float*)d_in[LB + L_CDIR], (const float*)d_in[LB + L_CUND],
        (const float*)d_in[LB + L_CALL], (const float*)d_in[LB + L_CONST],
        orig, res, h, hb, Cout);
  }

  // ---- decoder ----
  // z1 = relu(h @ W1^T + b1) -> bf16 only (chained into next WMMA)
  launch_gemm(hb, dW1, (const float*)d_in[IN_DEC_B1], nullptr, z1b, N, 64, 32, 1, stream);
  // logits = z1 @ W2^T + b2 (reuse agg[0] as logits buffer)
  launch_gemm(z1b, dW2, (const float*)d_in[IN_DEC_B2], agg[0], nullptr, N, 32, 32, 0, stream);

  float* out = (float*)d_out;
  logsoftmax32<<<(N + 7) / 8, 256, 0, stream>>>(agg[0], out, N);
  l2norm64<<<(N + 7) / 8, 256, 0, stream>>>(h, out + (size_t)N * 32, N);
  (void)out_size;
}
